// RiemannFMRelationSelfAttention_83141976916821
// MI455X (gfx1250) — matrix operand order repaired and checked by reading, not compiled, not run on hardware
//
#include <hip/hip_runtime.h>

// ---------------------------------------------------------------------------
// RiemannFM relation self-attention for MI455X (gfx1250, wave32, WMMA)
// B=32, K=512, REL_DIM=1024, H=16, D=64, REL_EMB_DIM=256
// ---------------------------------------------------------------------------

typedef __bf16 bf16_t;
typedef __attribute__((ext_vector_type(8)))  __bf16 v8bf;
typedef __attribute__((ext_vector_type(16))) __bf16 v16bf;
typedef __attribute__((ext_vector_type(8)))  float  v8f;

#define BATCH   32
#define SEQ     512
#define DMODEL  1024
#define NHEADS  16
#define HDIM    64
#define REMB    256
#define MROWS   (BATCH * SEQ)   // 16384
#define SCALE_F 0.125f          // 1/sqrt(64)

__device__ __forceinline__ v16bf cat16(v8bf lo, v8bf hi) {
  return __builtin_shufflevector(lo, hi, 0,1,2,3,4,5,6,7,8,9,10,11,12,13,14,15);
}

__device__ __forceinline__ v8bf cvt8(float4 a, float4 b) {
  v8bf r;
  r[0] = (__bf16)a.x; r[1] = (__bf16)a.y; r[2] = (__bf16)a.z; r[3] = (__bf16)a.w;
  r[4] = (__bf16)b.x; r[5] = (__bf16)b.y; r[6] = (__bf16)b.z; r[7] = (__bf16)b.w;
  return r;
}

// A-fragment (16x32 bf16): lane l -> row l%16; elems = K {8h..8h+7, 16+8h..16+8h+7}
__device__ __forceinline__ v16bf afrag(const __bf16* row, int lh) {
  v8bf lo = *(const v8bf*)(row + 8 * lh);
  v8bf hi = *(const v8bf*)(row + 16 + 8 * lh);
  return cat16(lo, hi);
}
// B-fragment (32x16 bf16), loaded from B^T rows: lane l -> col l%16; K {8h..8h+7, 16+8h..16+8h+7}
__device__ __forceinline__ v16bf bfrag(const __bf16* row, int lh) {
  v8bf lo = *(const v8bf*)(row + 8 * lh);
  v8bf hi = *(const v8bf*)(row + 16 + 8 * lh);
  return cat16(lo, hi);
}

__device__ __forceinline__ v8f wmma_bf16(v16bf a, v16bf b, v8f c) {
  return __builtin_amdgcn_wmma_f32_16x16x32_bf16(false, a, false, b, (short)0, c,
                                                 false, false);
}

// ---------------------------------------------------------------------------
// 1) inverse row norms of R (512 rows x 256)
// ---------------------------------------------------------------------------
__global__ void rnorm_kernel(const float* __restrict__ R, float* __restrict__ invn) {
  int i = blockIdx.x * blockDim.x + threadIdx.x;
  if (i < SEQ) {
    const float4* r4 = (const float4*)(R + (size_t)i * REMB);
    float s = 0.f;
#pragma unroll 8
    for (int t = 0; t < REMB / 4; ++t) {
      float4 x = r4[t];
      s += x.x * x.x + x.y * x.y + x.z * x.z + x.w * x.w;
    }
    invn[i] = 1.0f / fmaxf(sqrtf(s), 1e-12f);
  }
}

// ---------------------------------------------------------------------------
// 2) sim = R_norm @ R_norm^T  (512x512 f32) — computed once, reused B*H times
// ---------------------------------------------------------------------------
__global__ void __launch_bounds__(256) simmat_kernel(const float* __restrict__ R,
                                                     const float* __restrict__ invn,
                                                     float* __restrict__ simb) {
  int idx = blockIdx.x * 256 + threadIdx.x;
  int i = idx >> 9, j = idx & (SEQ - 1);
  const float4* a = (const float4*)(R + (size_t)i * REMB);
  const float4* b = (const float4*)(R + (size_t)j * REMB);
  float s = 0.f;
#pragma unroll 4
  for (int t = 0; t < REMB / 4; ++t) {
    float4 x = a[t], y = b[t];
    s += x.x * y.x + x.y * y.y + x.z * y.z + x.w * y.w;
  }
  simb[idx] = s * invn[i] * invn[j];
}

// ---------------------------------------------------------------------------
// 3) GEMM: C[m][n] = sum_k A[m][k] * W[n][k]   (y = x @ W^T, torch Linear)
//    M=16384, N=1024, K=1024.  Block: 256 thr (8 waves), tile 128(M)x128(N),
//    K staged 64-wide through LDS as bf16 -> 16 WMMAs per wave per barrier,
//    each A-fragment reused across 8 B-fragments.
//    A_F32: convert f32 A on the fly.  OUT_F32: f32 store (final GEMM).
//    TRANSV: store transposed (b,h,d,k) layout for the V projection.
// ---------------------------------------------------------------------------
template <bool A_F32, bool OUT_F32, bool TRANSV>
__global__ void __launch_bounds__(256) gemm_kernel(const void* __restrict__ Aptr,
                                                   const float* __restrict__ W,
                                                   void* __restrict__ Out) {
  const int nt = blockIdx.x, mt = blockIdx.y;
  const int tid = threadIdx.x;
  const int wave = tid >> 5, lane = tid & 31;
  const int lh = lane >> 4, ln = lane & 15;
  const int m0 = mt * 128, n0 = nt * 128;

  __shared__ __bf16 As[128][72];  // 128 x 64, +8 pad (bank-conflict-free frags)
  __shared__ __bf16 Bs[128][72];  // Bs[n][k] = W[n0+n][k0+k]

  const v8f vzero = {0.f, 0.f, 0.f, 0.f, 0.f, 0.f, 0.f, 0.f};
  v8f acc[8];
#pragma unroll
  for (int j = 0; j < 8; ++j) acc[j] = vzero;

  const int srow = tid >> 1;          // 0..127
  const int scol = (tid & 1) * 32;    // 0 or 32

  for (int k0 = 0; k0 < DMODEL; k0 += 64) {
    // ---- stage A tile (128 x 64) ----
    if constexpr (A_F32) {
      const float* A = (const float*)Aptr;
      const float4* src = (const float4*)(A + (size_t)(m0 + srow) * DMODEL + k0 + scol);
      *(v8bf*)&As[srow][scol]      = cvt8(src[0], src[1]);
      *(v8bf*)&As[srow][scol + 8]  = cvt8(src[2], src[3]);
      *(v8bf*)&As[srow][scol + 16] = cvt8(src[4], src[5]);
      *(v8bf*)&As[srow][scol + 24] = cvt8(src[6], src[7]);
      if (k0 + 64 < DMODEL)
        __builtin_prefetch((const char*)(src) + 64 * sizeof(float), 0, 0);
    } else {
      const __bf16* A = (const __bf16*)Aptr;
      const uint4* src = (const uint4*)(A + (size_t)(m0 + srow) * DMODEL + k0 + scol);
      *(uint4*)&As[srow][scol]      = src[0];
      *(uint4*)&As[srow][scol + 8]  = src[1];
      *(uint4*)&As[srow][scol + 16] = src[2];
      *(uint4*)&As[srow][scol + 24] = src[3];
      if (k0 + 64 < DMODEL)
        __builtin_prefetch((const char*)(src) + 64 * sizeof(__bf16), 0, 0);
    }
    // ---- stage B tile (128 x 64) from fp32 weights ----
    {
      const float4* src = (const float4*)(W + (size_t)(n0 + srow) * DMODEL + k0 + scol);
      *(v8bf*)&Bs[srow][scol]      = cvt8(src[0], src[1]);
      *(v8bf*)&Bs[srow][scol + 8]  = cvt8(src[2], src[3]);
      *(v8bf*)&Bs[srow][scol + 16] = cvt8(src[4], src[5]);
      *(v8bf*)&Bs[srow][scol + 24] = cvt8(src[6], src[7]);
      if (k0 + 64 < DMODEL)
        __builtin_prefetch((const char*)(src) + 64 * sizeof(float), 0, 0);
    }
    __syncthreads();

    // ---- 16 WMMAs per wave: 2 K-subchunks x 8 N-subtiles ----
    const __bf16* arowp = &As[wave * 16 + ln][0];
#pragma unroll
    for (int c = 0; c < 2; ++c) {
      v16bf a = afrag(arowp + 32 * c, lh);
#pragma unroll
      for (int j = 0; j < 8; ++j) {
        v16bf b = bfrag(&Bs[j * 16 + ln][32 * c], lh);
        acc[j] = wmma_bf16(a, b, acc[j]);
      }
    }
    __syncthreads();
  }

#pragma unroll
  for (int j = 0; j < 8; ++j) {
#pragma unroll
    for (int r = 0; r < 8; ++r) {
      const int gm = m0 + wave * 16 + r + 8 * lh;   // global row (b*512 + k)
      const int gn = n0 + j * 16 + ln;              // global col (h*64 + d)
      const float val = acc[j][r];
      if constexpr (OUT_F32) {
        ((float*)Out)[(size_t)gm * DMODEL + gn] = val;
      } else if constexpr (TRANSV) {
        const int bb = gm >> 9, kk = gm & (SEQ - 1);
        ((__bf16*)Out)[((size_t)bb * DMODEL + gn) * SEQ + kk] = (__bf16)val;
      } else {
        ((__bf16*)Out)[(size_t)gm * DMODEL + gn] = (__bf16)val;
      }
    }
  }
}

// ---------------------------------------------------------------------------
// 4) Flash attention per (b,h): S = q k^T * scale + w_sim[h]*sim, online
//    softmax, O = P v.  Block: 128 thr (4 waves), each wave owns a 16-row
//    query tile; K/V chunks of 32 keys staged in LDS shared by all waves.
// ---------------------------------------------------------------------------
__global__ void __launch_bounds__(128) attn_kernel(
    const __bf16* __restrict__ q, const __bf16* __restrict__ k,
    const __bf16* __restrict__ vT, const float* __restrict__ sim,
    const float* __restrict__ w_sim, __bf16* __restrict__ out) {
  const int b = blockIdx.z, h = blockIdx.y, qb = blockIdx.x;
  const int tid = threadIdx.x, wave = tid >> 5, lane = tid & 31;
  const int lh = lane >> 4, ln = lane & 15;
  const int q0 = qb * 64 + wave * 16;  // query tile start within SEQ

  __shared__ __bf16 Ks[32][72];        // 32 keys x 64 d   (pad 8)
  __shared__ __bf16 Vs[64][40];        // 64 d   x 32 keys (pad 8)
  __shared__ __bf16 Ps[4][16][40];     // per-wave P re-shuffle scratch

  const v8f vzero = {0.f, 0.f, 0.f, 0.f, 0.f, 0.f, 0.f, 0.f};

  // Q A-fragments for the two K-chunks of the d=64 inner dimension
  v16bf qa[2];
  {
    const __bf16* qrow = q + ((size_t)(b * SEQ + q0 + ln) * DMODEL + h * HDIM);
#pragma unroll
    for (int c = 0; c < 2; ++c) qa[c] = afrag(qrow + 32 * c, lh);
  }

  float mi[8], li[8];
#pragma unroll
  for (int r = 0; r < 8; ++r) { mi[r] = -1e30f; li[r] = 0.f; }
  v8f O[4];
#pragma unroll
  for (int j = 0; j < 4; ++j) O[j] = vzero;

  const float wsim = w_sim[h];
  const __bf16* kbase = k + ((size_t)(b * SEQ) * DMODEL + h * HDIM);
  const __bf16* vbase = vT + ((size_t)(b * DMODEL + h * HDIM)) * SEQ;

  for (int kt = 0; kt < SEQ / 32; ++kt) {
    {  // stage K chunk: 32 keys x 64 d
      const int row = tid >> 2, col = (tid & 3) * 16;
      const uint4* src = (const uint4*)(kbase + (size_t)(kt * 32 + row) * DMODEL + col);
      *(uint4*)&Ks[row][col]     = src[0];
      *(uint4*)&Ks[row][col + 8] = src[1];
    }
    {  // stage V chunk (transposed layout): 64 d x 32 keys
      const int row = tid >> 1, col = (tid & 1) * 16;
      const uint4* src = (const uint4*)(vbase + (size_t)row * SEQ + kt * 32 + col);
      *(uint4*)&Vs[row][col]     = src[0];
      *(uint4*)&Vs[row][col + 8] = src[1];
    }
    __syncthreads();

    // scores for two 16-key subtiles
    float sv[2][8];
#pragma unroll
    for (int s = 0; s < 2; ++s) {
      v8f sc = vzero;
#pragma unroll
      for (int c = 0; c < 2; ++c)
        sc = wmma_bf16(qa[c], bfrag(&Ks[s * 16 + ln][32 * c], lh), sc);
#pragma unroll
      for (int r = 0; r < 8; ++r) {
        const float bias =
            sim[(size_t)(q0 + r + 8 * lh) * SEQ + kt * 32 + s * 16 + ln];
        sv[s][r] = sc[r] * SCALE_F + wsim * bias;
      }
    }

    // online softmax (rows live across 16-lane halves)
#pragma unroll
    for (int r = 0; r < 8; ++r) {
      float mx = fmaxf(sv[0][r], sv[1][r]);
      mx = fmaxf(mx, __shfl_xor(mx, 1, 32));
      mx = fmaxf(mx, __shfl_xor(mx, 2, 32));
      mx = fmaxf(mx, __shfl_xor(mx, 4, 32));
      mx = fmaxf(mx, __shfl_xor(mx, 8, 32));
      const float mnew = fmaxf(mi[r], mx);
      const float corr = __expf(mi[r] - mnew);
      const float p0 = __expf(sv[0][r] - mnew);
      const float p1 = __expf(sv[1][r] - mnew);
      float ps = p0 + p1;
      ps += __shfl_xor(ps, 1, 32);
      ps += __shfl_xor(ps, 2, 32);
      ps += __shfl_xor(ps, 4, 32);
      ps += __shfl_xor(ps, 8, 32);
      li[r] = li[r] * corr + ps;
      mi[r] = mnew;
      sv[0][r] = p0; sv[1][r] = p1;
#pragma unroll
      for (int j = 0; j < 4; ++j) O[j][r] *= corr;
    }

    // C-layout -> A-layout via per-wave LDS scratch
#pragma unroll
    for (int r = 0; r < 8; ++r) {
      Ps[wave][r + 8 * lh][ln]      = (__bf16)sv[0][r];
      Ps[wave][r + 8 * lh][16 + ln] = (__bf16)sv[1][r];
    }
    v16bf pa = afrag(&Ps[wave][ln][0], lh);

    // O += P @ V  (B^T rows contiguous thanks to transposed V layout)
#pragma unroll
    for (int j = 0; j < 4; ++j)
      O[j] = wmma_bf16(pa, bfrag(&Vs[j * 16 + ln][0], lh), O[j]);

    __syncthreads();
  }

  // normalize + store bf16 (row-major (b*K+k, h*64+d) for the final GEMM)
#pragma unroll
  for (int r = 0; r < 8; ++r) {
    const float inv = 1.0f / li[r];
    const int gq = b * SEQ + q0 + r + 8 * lh;
#pragma unroll
    for (int j = 0; j < 4; ++j)
      out[(size_t)gq * DMODEL + h * HDIM + j * 16 + ln] = (__bf16)(O[j][r] * inv);
  }
}

// ---------------------------------------------------------------------------
// Launch: sim precompute -> q/k/v projections -> flash attention -> W_o GEMM
// inputs: 0:h_R 1:R 2:W_q 3:W_k 4:W_v 5:W_o 6:w_sim
// ---------------------------------------------------------------------------
extern "C" void kernel_launch(void* const* d_in, const int* in_sizes, int n_in,
                              void* d_out, int out_size, void* d_ws, size_t ws_size,
                              hipStream_t stream) {
  const float* h_R  = (const float*)d_in[0];
  const float* R    = (const float*)d_in[1];
  const float* W_q  = (const float*)d_in[2];
  const float* W_k  = (const float*)d_in[3];
  const float* W_v  = (const float*)d_in[4];
  const float* W_o  = (const float*)d_in[5];
  const float* wsim = (const float*)d_in[6];

  char* ws = (char*)d_ws;
  const size_t SZ = (size_t)MROWS * DMODEL * sizeof(__bf16);  // 32 MiB each
  __bf16* qbuf = (__bf16*)(ws);
  __bf16* kbuf = (__bf16*)(ws + SZ);
  __bf16* vbuf = (__bf16*)(ws + 2 * SZ);
  __bf16* obuf = (__bf16*)(ws + 3 * SZ);
  float*  simb = (float*)(ws + 4 * SZ);
  float*  invn = (float*)(ws + 4 * SZ + (size_t)SEQ * SEQ * sizeof(float));

  rnorm_kernel<<<2, 256, 0, stream>>>(R, invn);
  simmat_kernel<<<(SEQ * SEQ) / 256, 256, 0, stream>>>(R, invn, simb);

  dim3 gg(DMODEL / 128, MROWS / 128);  // (8, 128)
  gemm_kernel<true, false, false><<<gg, 256, 0, stream>>>(h_R, W_q, qbuf);
  gemm_kernel<true, false, false><<<gg, 256, 0, stream>>>(h_R, W_k, kbuf);
  gemm_kernel<true, false, true ><<<gg, 256, 0, stream>>>(h_R, W_v, vbuf);

  attn_kernel<<<dim3(SEQ / 64, NHEADS, BATCH), 128, 0, stream>>>(
      qbuf, kbuf, vbuf, simb, wsim, obuf);

  gemm_kernel<false, true, false><<<gg, 256, 0, stream>>>(obuf, W_o, d_out);
}
